// PathFinder128Net_33973191311910
// MI455X (gfx1250) — matrix operand order repaired and checked by reading, not compiled, not run on hardware
//
#include <hip/hip_runtime.h>
#include <hip/hip_bf16.h>
#include <math.h>

// ---------------------------------------------------------------------------
// PathFinder128Net forward -> final[:, 0].  Convs + FC heads as f16 WMMA
// GEMMs (f32 accumulate).  One wave owns a 16-pixel tile and computes ALL
// Cout/16 output tiles for it: the activation gather (A fragment) is built
// once per cin-chunk and reused by CT WMMAs, each adding just 2 x b128
// weight loads.  K padded to 32 per input channel -> guard-free chunks.
// ---------------------------------------------------------------------------

typedef __attribute__((ext_vector_type(16))) _Float16 v16h;
typedef __attribute__((ext_vector_type(8)))  _Float16 v8h;
typedef __attribute__((ext_vector_type(8)))  float    v8f;

union V16H { v16h v; v8h h8[2]; _Float16 e[16]; };
union V8F  { v8f  v; float e[8]; };

#define BATCH 64

// ---------------- f32 -> f16 cast ----------------
__global__ void cast_f32_f16(const float* __restrict__ in, _Float16* __restrict__ out, int n) {
    int i = blockIdx.x * blockDim.x + threadIdx.x;
    if (i < n) out[i] = (_Float16)in[i];
}

// ------- conv weight pack: OIHW f32 [Cout][Cin][25] -> f16 [Cout][Cin][32], taps 25..31 = 0
__global__ void pack_conv_w(const float* __restrict__ w, _Float16* __restrict__ wp, int total) {
    int i = blockIdx.x * blockDim.x + threadIdx.x;     // total = Cout*Cin*32
    if (i >= total) return;
    int t  = i & 31;
    int cc = i >> 5;                                   // cout*Cin + cin
    wp[i] = (t < 25) ? (_Float16)w[cc * 25 + t] : (_Float16)0.0f;
}

// ------- FC weight transpose-cast: f32 [K][N] -> f16 [N][K]
__global__ void transpose_fc_w(const float* __restrict__ w, _Float16* __restrict__ wt,
                               int K, int N) {
    int i = blockIdx.x * blockDim.x + threadIdx.x;
    if (i >= K * N) return;
    int k   = i % K;
    int col = i / K;
    wt[(size_t)col * K + k] = (_Float16)w[(size_t)k * N + col];
}

// ---------------- 4x4 max pool (VALID, stride 4) ----------------
__global__ void maxpool4(const _Float16* __restrict__ in, _Float16* __restrict__ out,
                         int C, int H, int W, int Ho, int Wo, int n) {
    int i = blockIdx.x * blockDim.x + threadIdx.x;
    if (i >= n) return;
    int x = i % Wo; int t = i / Wo;
    int y = t % Ho; t /= Ho;
    int c = t % C;  int b = t / C;
    const _Float16* p = in + (((size_t)b * C + c) * H + y * 4) * W + x * 4;
    float m = -INFINITY;
#pragma unroll
    for (int dy = 0; dy < 4; ++dy)
#pragma unroll
        for (int dx = 0; dx < 4; ++dx) {
            float v = (float)p[dy * W + dx];
            m = v > m ? v : m;
        }
    out[i] = (_Float16)m;
}

// ---------------- implicit-GEMM conv via WMMA ----------------
// Wave = one 16-pixel tile x all CT cout-tiles.  A = activation gather
// (M = pixel, built once per cin), B = packed weights [Cout][CIN*32]
// streamed as 2 x b128 per (cin, cout-tile).
template <int CIN, int CT>
__global__ void conv_wmma(const _Float16* __restrict__ act,   // [B][CIN][H][W]
                          const _Float16* __restrict__ wp,    // [CT*16][CIN*32]
                          const float*    __restrict__ bias,
                          _Float16*       __restrict__ out,   // [B][CT*16][Ho][Wo]
                          int H, int W, int Ho, int Wo) {
    const int  lane   = threadIdx.x & 31;
    const int  wave   = (blockIdx.x * blockDim.x + threadIdx.x) >> 5;
    const int  Ntot   = BATCH * Ho * Wo;               // pixels (GEMM M)
    const int  pTiles = (Ntot + 15) >> 4;
    if (wave >= pTiles) return;                        // wave-uniform -> EXEC all-1s
    const int  p0     = wave << 4;
    const int  laneHi = lane >> 4;
    const int  laneLo = lane & 15;
    const int  Cout   = CT * 16;
    const int  Kpad   = CIN * 32;

    // A-fragment pixel: lane holds M = laneLo for both half-K groups.
    const int  pa      = p0 + laneLo;
    const bool paValid = pa < Ntot;
    const int  pac = paValid ? pa : 0;
    const int  aox = pac % Wo;
    int        t   = pac / Wo;
    const int  aoy = t % Ho;
    const int  ab  = t / Ho;
    const _Float16* actA = act + ((size_t)ab * CIN * H + aoy) * W + aox;

    // B-fragment weight rows: N = laneLo + 16*ct, K = h + 16*laneHi (contiguous).
    const _Float16* wrow = wp + (size_t)laneLo * Kpad + (laneHi << 4);

    V8F c[CT];
#pragma unroll
    for (int ct = 0; ct < CT; ++ct)
#pragma unroll
        for (int r = 0; r < 8; ++r) c[ct].e[r] = 0.0f;

#pragma unroll
    for (int cin = 0; cin < CIN; ++cin) {
        V16H a;
        const _Float16* ap = actA + (size_t)cin * H * W;
#pragma unroll
        for (int h = 0; h < 16; ++h) {
            // A layout: K = h + 16*(lane>>4); chunk-local tap < 32
            int tap = h + (laneHi << 4);
            _Float16 v = (_Float16)0.0f;
            if (tap < 25 && paValid) {
                int dy = tap / 5;
                int dx = tap - dy * 5;
                v = ap[dy * W + dx];
            }
            a.e[h] = v;
        }
#pragma unroll
        for (int ct = 0; ct < CT; ++ct) {
            V16H b;
            const _Float16* wchunk = wrow + (size_t)(ct * 16) * Kpad + cin * 32;
            b.h8[0] = *(const v8h*)(wchunk);
            b.h8[1] = *(const v8h*)(wchunk + 8);
            c[ct].v = __builtin_amdgcn_wmma_f32_16x16x32_f16(
                false, a.v, false, b.v, (short)0, c[ct].v, false, false);
        }
    }

    // Epilogue: D VGPR r -> pixel M = r + 8*laneHi, cout N = laneLo + 16*ct.
#pragma unroll
    for (int r = 0; r < 8; ++r) {
        int p = p0 + (laneHi << 3) + r;
        if (p < Ntot) {
            int ox = p % Wo;
            int tt = p / Wo;
            int oy = tt % Ho;
            int bb = tt / Ho;
            _Float16* orow = out + ((size_t)bb * Cout * Ho + oy) * Wo + ox;
#pragma unroll
            for (int ct = 0; ct < CT; ++ct) {
                int coutN = ct * 16 + laneLo;
                orow[(size_t)coutN * Ho * Wo] = (_Float16)(c[ct].e[r] + bias[coutN]);
            }
        }
    }
}

// ---------------- FC head: relu(emb[64,1600] @ W[1600,256] + bias) ----------------
// Both fragments fully vectorized: emb rows and wt[N][K] rows are K-contiguous.
__global__ void fc_relu_wmma(const _Float16* __restrict__ emb,  // [64][1600]
                             const _Float16* __restrict__ wt,   // [256][1600]
                             const float*    __restrict__ bias,
                             float*          __restrict__ hout) {
    const int lane = threadIdx.x & 31;
    const int wave = (blockIdx.x * blockDim.x + threadIdx.x) >> 5;
    const int Mt = 4, Nt = 16;                          // 64x256 output, 64 waves
    if (wave >= Mt * Nt) return;
    const int mT = wave % Mt;
    const int nT = wave / Mt;
    const int m0 = mT << 4;
    const int n0 = nT << 4;
    const int laneHi = lane >> 4;
    const int laneLo = lane & 15;

    const _Float16* arow = emb + (size_t)(m0 + laneLo) * 1600;
    const _Float16* brow = wt  + (size_t)(n0 + laneLo) * 1600 + (laneHi << 4);

    V8F c;
#pragma unroll
    for (int r = 0; r < 8; ++r) c.e[r] = 0.0f;

    for (int kbase = 0; kbase < 1600; kbase += 32) {
        V16H a, b;
        // A: halfs 0-15 -> K = kbase + 16*laneHi + 0..15
        a.h8[0] = *(const v8h*)(arow + kbase + (laneHi << 4));
        a.h8[1] = *(const v8h*)(arow + kbase + 8 + (laneHi << 4));
        // B: halfs 0-15 -> K = kbase + 16*laneHi + 0..15 (folded into brow)
        b.h8[0] = *(const v8h*)(brow + kbase);
        b.h8[1] = *(const v8h*)(brow + kbase + 8);
        c.v = __builtin_amdgcn_wmma_f32_16x16x32_f16(
            false, a.v, false, b.v, (short)0, c.v, false, false);
    }

    const int   col = n0 + laneLo;
    const float bv  = bias[col];
#pragma unroll
    for (int r = 0; r < 8; ++r) {
        int row = m0 + r + (laneHi << 3);
        float v = c.e[r] + bv;
        hout[row * 256 + col] = v > 0.0f ? v : 0.0f;
    }
}

// ---------------- final: sigmoid dots + product ----------------
__global__ void head_final(const float* __restrict__ h_ed,
                           const float* __restrict__ h_ep,
                           const float* __restrict__ w_ed2, const float* __restrict__ b_ed2,
                           const float* __restrict__ w_ep2, const float* __restrict__ b_ep2,
                           const int*   __restrict__ edge_src, const int* __restrict__ edge_dst,
                           float* __restrict__ out) {
    int b = threadIdx.x;
    if (b >= BATCH) return;
    const int s0 = edge_src[0];                        // (0, 6) for the 6x6 grid
    const int d0 = edge_dst[0];
    float sE = 0.0f, sS = 0.0f, sD = 0.0f;
    for (int j = 0; j < 256; ++j) {
        float he = h_ed[b * 256 + j];
        float hp = h_ep[b * 256 + j];
        sE += he * w_ed2[j * 120 + 0];
        sS += hp * w_ep2[j * 36 + s0];
        sD += hp * w_ep2[j * 36 + d0];
    }
    float E  = 1.0f / (1.0f + expf(-(sE + b_ed2[0])));
    float Ps = 1.0f / (1.0f + expf(-(sS + b_ep2[s0])));
    float Pd = 1.0f / (1.0f + expf(-(sD + b_ep2[d0])));
    out[b] = E * Ps * Pd;
}

// ---------------------------------------------------------------------------
extern "C" void kernel_launch(void* const* d_in, const int* in_sizes, int n_in,
                              void* d_out, int out_size, void* d_ws, size_t ws_size,
                              hipStream_t stream) {
    const float* x    = (const float*)d_in[0];
    const float* w1   = (const float*)d_in[1];
    const float* b1   = (const float*)d_in[2];
    const float* w2   = (const float*)d_in[3];
    const float* b2   = (const float*)d_in[4];
    const float* w3   = (const float*)d_in[5];
    const float* b3   = (const float*)d_in[6];
    const float* w4   = (const float*)d_in[7];
    const float* b4   = (const float*)d_in[8];
    const float* wed1 = (const float*)d_in[9];
    const float* bed1 = (const float*)d_in[10];
    const float* wed2 = (const float*)d_in[11];
    const float* bed2 = (const float*)d_in[12];
    const float* wep1 = (const float*)d_in[13];
    const float* bep1 = (const float*)d_in[14];
    const float* wep2 = (const float*)d_in[15];
    const float* bep2 = (const float*)d_in[16];
    const int*   esrc = (const int*)d_in[17];
    const int*   edst = (const int*)d_in[18];
    float* out = (float*)d_out;

    // ---- workspace carving (256B aligned) ----
    char* ws = (char*)d_ws;
    size_t off = 0;
    auto take = [&](size_t bytes) -> void* {
        void* p = ws + off;
        off = (off + bytes + 255) & ~(size_t)255;
        return p;
    };
    _Float16* xh    = (_Float16*)take((size_t)BATCH * 128 * 128 * 2);
    _Float16* w1p   = (_Float16*)take(32 * 1  * 32 * 2);
    _Float16* w2p   = (_Float16*)take(32 * 32 * 32 * 2);
    _Float16* w3p   = (_Float16*)take(64 * 32 * 32 * 2);
    _Float16* w4p   = (_Float16*)take(64 * 64 * 32 * 2);
    _Float16* wed1t = (_Float16*)take((size_t)256 * 1600 * 2);
    _Float16* wep1t = (_Float16*)take((size_t)256 * 1600 * 2);
    _Float16* act1  = (_Float16*)take((size_t)BATCH * 32 * 124 * 124 * 2);
    _Float16* act2  = (_Float16*)take((size_t)BATCH * 32 * 120 * 120 * 2);
    _Float16* pool2 = (_Float16*)take((size_t)BATCH * 32 * 30 * 30 * 2);
    _Float16* act3  = (_Float16*)take((size_t)BATCH * 64 * 26 * 26 * 2);
    _Float16* act4  = (_Float16*)take((size_t)BATCH * 64 * 22 * 22 * 2);
    _Float16* embh  = (_Float16*)take((size_t)BATCH * 1600 * 2);
    float*    h_ed  = (float*)take((size_t)BATCH * 256 * 4);
    float*    h_ep  = (float*)take((size_t)BATCH * 256 * 4);

    {   // input cast
        int n = BATCH * 128 * 128;
        cast_f32_f16<<<(n + 255) / 256, 256, 0, stream>>>(x, xh, n);
    }
    auto pack = [&](const float* src, _Float16* dst, int coutCin) {
        int n = coutCin * 32;
        pack_conv_w<<<(n + 255) / 256, 256, 0, stream>>>(src, dst, n);
    };
    pack(w1, w1p, 32 * 1);
    pack(w2, w2p, 32 * 32);
    pack(w3, w3p, 64 * 32);
    pack(w4, w4p, 64 * 64);
    {
        int n = 1600 * 256;
        transpose_fc_w<<<(n + 255) / 256, 256, 0, stream>>>(wed1, wed1t, 1600, 256);
        transpose_fc_w<<<(n + 255) / 256, 256, 0, stream>>>(wep1, wep1t, 1600, 256);
    }

    auto blocksFor = [&](int H, int W) {
        int Ho = H - 4, Wo = W - 4;
        int pTiles = (BATCH * Ho * Wo + 15) / 16;
        return (pTiles + 7) / 8;                       // 8 waves / 256-thread block
    };

    // conv1: [64,1,128,128] -> [64,32,124,124]
    conv_wmma<1, 2><<<blocksFor(128, 128), 256, 0, stream>>>(
        xh, w1p, b1, act1, 128, 128, 124, 124);
    // conv2: -> [64,32,120,120]
    conv_wmma<32, 2><<<blocksFor(124, 124), 256, 0, stream>>>(
        act1, w2p, b2, act2, 124, 124, 120, 120);
    {
        int n = BATCH * 32 * 30 * 30;
        maxpool4<<<(n + 255) / 256, 256, 0, stream>>>(act2, pool2, 32, 120, 120, 30, 30, n);
    }
    // conv3: [64,32,30,30] -> [64,64,26,26]
    conv_wmma<32, 4><<<blocksFor(30, 30), 256, 0, stream>>>(
        pool2, w3p, b3, act3, 30, 30, 26, 26);
    // conv4: -> [64,64,22,22]
    conv_wmma<64, 4><<<blocksFor(26, 26), 256, 0, stream>>>(
        act3, w4p, b4, act4, 26, 26, 22, 22);
    {
        int n = BATCH * 64 * 5 * 5;                    // emb order == [b][c][y][x]
        maxpool4<<<(n + 255) / 256, 256, 0, stream>>>(act4, embh, 64, 22, 22, 5, 5, n);
    }

    fc_relu_wmma<<<8, 256, 0, stream>>>(embh, wed1t, bed1, h_ed);
    fc_relu_wmma<<<8, 256, 0, stream>>>(embh, wep1t, bep1, h_ep);

    head_final<<<1, 64, 0, stream>>>(h_ed, h_ep, wed2, bed2, wep2, bep2, esrc, edst, out);
}
